// SharedMoEBlock_18502719111702
// MI455X (gfx1250) — matrix operand order repaired and checked by reading, not compiled, not run on hardware
//
#include <hip/hip_runtime.h>
#include <hip/hip_bf16.h>
#include <cstdint>
#include <cstddef>

// Problem sizes (fixed by the reference).
#define B_  4
#define S_  2048
#define T_  (B_ * S_)   // 8192 tokens
#define D_  1024
#define H_  2048
#define E_  8

// ---- CDNA5 WMMA types -------------------------------------------------------
typedef __attribute__((ext_vector_type(16))) __bf16       v16bf;
typedef __attribute__((ext_vector_type(8)))  float        v8f;
typedef __attribute__((ext_vector_type(4)))  unsigned int u32x4;
typedef __attribute__((ext_vector_type(8)))  unsigned int u32x8;
typedef __attribute__((ext_vector_type(4)))  float        f32x4;

union FragBF { v16bf v; u32x4 u[2]; };

// GEMM tiling: 128x128 block tile, K-step 32, 256 threads = 8 wave32.
#define TM   128
#define TN   128
#define TK   32
#define LSTR 40   // LDS row stride in bf16 (32 + 8 pad): rows stay 16B-aligned

__device__ __forceinline__ v8f wmma_bf16(v16bf a, v16bf b, v8f c) {
  // v_wmma_f32_16x16x32_bf16  (neg_a, A, neg_b, B, c_mod, C, reuse_a, reuse_b)
  return __builtin_amdgcn_wmma_f32_16x16x32_bf16(false, a, false, b, (short)0, c,
                                                 false, false);
}

// Async global->LDS 16B copy (CDNA5 GLOBAL_LOAD_ASYNC_TO_LDS_B128, ASYNCcnt).
// LDS dest address = low 32 bits of the generic pointer (flat LDS aperture).
__device__ __forceinline__ void async_copy_b128(const void* gsrc, void* lds_dst) {
  const unsigned lds_off = (unsigned)(unsigned long long)lds_dst;
  asm volatile("global_load_async_to_lds_b128 %0, %1, off"
               :: "v"(lds_off), "v"(gsrc)
               : "memory");
}
__device__ __forceinline__ void wait_async0() {
  asm volatile("s_wait_asynccnt 0" ::: "memory");
}

// ---- Tensor Data Mover: 2D bf16 tile, global -> LDS (ISA 08, D# groups 0/1) -
// Loads a (tile_rows x TK) bf16 tile whose rows are `stride_elems` apart,
// zero-filling rows past `rows_valid` (OOB reads return 0), and padding the
// LDS destination 16B after every 64B row -> exactly our LSTR=40 layout.
// One issue per wave (TDM ignores EXEC); caller guards to a single wave.
__device__ __forceinline__ void tdm_load_tile(unsigned lds_addr, const void* gaddr,
                                              unsigned rows_valid,
                                              unsigned tile_rows,
                                              unsigned stride_elems) {
  const unsigned long long ga = (unsigned long long)gaddr;
  u32x4 g0;
  g0.x = 1u;                                   // count=1 user descriptor
  g0.y = lds_addr;                             // lds_addr [63:32]
  g0.z = (unsigned)ga;                         // global_addr [95:64]
  g0.w = (unsigned)(ga >> 32) | (2u << 30);    // global_addr[56:32] | type=2
  u32x8 g1;
  // data_size=1 (2B) | pad_enable | pad_interval=3 (16 dw) | pad_amount=3 (4 dw)
  g1.s0 = (1u << 16) | (1u << 20) | (3u << 22) | (3u << 25);
  const unsigned td0 = stride_elems;           // dim0 extent (full rows)
  g1.s1 = (td0 & 0xFFFFu) << 16;               // [63:48]   tensor_dim0 lo
  g1.s2 = (td0 >> 16) | ((rows_valid & 0xFFFFu) << 16);  // dim0 hi | dim1 lo
  g1.s3 = (rows_valid >> 16) | ((unsigned)TK << 16);     // dim1 hi | tile_dim0
  g1.s4 = tile_rows & 0xFFFFu;                 // tile_dim1 | tile_dim2=0
  g1.s5 = stride_elems;                        // tensor_dim0_stride [31:0]
  g1.s6 = 0u;                                  // stride hi16 | dim1_stride lo16
  g1.s7 = 0u;
  asm volatile("tensor_load_to_lds %0, %1" :: "s"(g0), "s"(g1) : "memory");
}

// A-matrix 16x32 bf16 fragment (ISA 7.12.2): lane<16 holds M=lane, K={0..7,16..23};
// lane>=16 holds M=lane-16, K={8..15,24..31}. Two 16B LDS chunks per lane.
__device__ __forceinline__ v16bf load_a_frag(const __bf16* As, int m0, int lane) {
  const int half = lane >> 4;
  const int m    = m0 + (lane & 15);
  const __bf16* row = As + m * LSTR;
  FragBF f;
  f.u[0] = *(const u32x4*)(row + half * 8);        // K = half*8 + 0..7
  f.u[1] = *(const u32x4*)(row + 16 + half * 8);   // K = 16 + half*8 + 0..7
  return f.v;
}

// B-matrix 32x16 bf16 fragment: lane holds N=lane&15, K = {h*8+0..7, 16+h*8+0..7}
// (h = lane>>4, same K order as A). Bs stores W[n][k] row-major (W is [N,K] = B^T).
__device__ __forceinline__ v16bf load_b_frag(const __bf16* Bs, int n0, int lane) {
  const int half = lane >> 4;
  const __bf16* row = Bs + (n0 + (lane & 15)) * LSTR + half * 8;
  FragBF f;
  f.u[0] = *(const u32x4*)(row);
  f.u[1] = *(const u32x4*)(row + 16);
  return f.v;
}

// fp32x16 (4 x f32x4 regs) -> bf16x16 -> two 16B LDS stores.
__device__ __forceinline__ void cvt_store16(__bf16* dst, const f32x4* r) {
  alignas(16) __bf16 tmp[16];
#pragma unroll
  for (int i = 0; i < 4; ++i) {
    tmp[4 * i + 0] = (__bf16)r[i].x; tmp[4 * i + 1] = (__bf16)r[i].y;
    tmp[4 * i + 2] = (__bf16)r[i].z; tmp[4 * i + 3] = (__bf16)r[i].w;
  }
  ((u32x4*)dst)[0] = ((const u32x4*)tmp)[0];
  ((u32x4*)dst)[1] = ((const u32x4*)tmp)[1];
}

// ---- small utility kernels --------------------------------------------------
__global__ void zero_f32_kernel(float* p, size_t n) {
  for (size_t i = blockIdx.x * (size_t)blockDim.x + threadIdx.x; i < n;
       i += (size_t)gridDim.x * blockDim.x)
    p[i] = 0.0f;
}

__global__ void zero_i32_kernel(int* p, int n) {
  int i = blockIdx.x * blockDim.x + threadIdx.x;
  if (i < n) p[i] = 0;
}

__global__ void cvt_bf16_kernel(const float* __restrict__ x,
                                unsigned short* __restrict__ out, size_t n) {
  __bf16* o = (__bf16*)out;
  for (size_t i = blockIdx.x * (size_t)blockDim.x + threadIdx.x; i < n;
       i += (size_t)gridDim.x * blockDim.x)
    o[i] = (__bf16)x[i];
}

__global__ void scale_kernel(float* __restrict__ out,
                             const float* __restrict__ ls, size_t n) {
  for (size_t i = blockIdx.x * (size_t)blockDim.x + threadIdx.x; i < n;
       i += (size_t)gridDim.x * blockDim.x)
    out[i] *= ls[i & (D_ - 1)];
}

// ---- router: softmax + top-2 + renorm, builds per-expert gather lists -------
__global__ __launch_bounds__(256)
void router_kernel(const float* __restrict__ x, const float* __restrict__ wr,
                   int* __restrict__ counts, int* __restrict__ idx,
                   float* __restrict__ wgt) {
  const int lane = threadIdx.x & 31;
  const int wave = threadIdx.x >> 5;
  const int t    = blockIdx.x * 8 + wave;
  if (t >= T_) return;

  float acc[E_];
#pragma unroll
  for (int e = 0; e < E_; ++e) acc[e] = 0.0f;

  const float* xrow = x + (size_t)t * D_;
  for (int d = lane; d < D_; d += 32) {
    const float xv = xrow[d];
#pragma unroll
    for (int e = 0; e < E_; ++e) acc[e] += xv * wr[e * D_ + d];
  }
#pragma unroll
  for (int e = 0; e < E_; ++e) {
#pragma unroll
    for (int off = 16; off > 0; off >>= 1) acc[e] += __shfl_xor(acc[e], off, 32);
  }
  if (lane == 0) {
    float mx = acc[0];
#pragma unroll
    for (int e = 1; e < E_; ++e) mx = fmaxf(mx, acc[e]);
    float p[E_]; float sum = 0.0f;
#pragma unroll
    for (int e = 0; e < E_; ++e) { p[e] = __expf(acc[e] - mx); sum += p[e]; }
    const float inv = 1.0f / sum;
#pragma unroll
    for (int e = 0; e < E_; ++e) p[e] *= inv;
    // top-2 (ties -> lowest index, matching jax.lax.top_k)
    int i1 = 0;
#pragma unroll
    for (int e = 1; e < E_; ++e) if (p[e] > p[i1]) i1 = e;
    int i2 = (i1 == 0) ? 1 : 0;
#pragma unroll
    for (int e = 0; e < E_; ++e) if (e != i1 && p[e] > p[i2]) i2 = e;
    const float rn = 1.0f / (p[i1] + p[i2]);
    int pos1 = atomicAdd(&counts[i1], 1);
    idx[i1 * T_ + pos1] = t;  wgt[i1 * T_ + pos1] = p[i1] * rn;
    int pos2 = atomicAdd(&counts[i2], 1);
    idx[i2 * T_ + pos2] = t;  wgt[i2 * T_ + pos2] = p[i2] * rn;
  }
}

// ---- gathered gate+up GEMM, async A staging, fused SiLU*up*weight epilogue --
// act[r, h] = bf16( silu(x[idx[r]] . wg[h]) * (x[idx[r]] . wu[h]) * wgt[r] )
// idx==nullptr -> identity gather over all T_ tokens with weight 1 (shared).
// A rows are gathered (non-contiguous) -> per-lane async-to-LDS staging.
__global__ __launch_bounds__(256)
void gate_up_kernel(const unsigned short* __restrict__ xb_,
                    const float* __restrict__ wg, const float* __restrict__ wu,
                    unsigned short* __restrict__ act_,
                    const int* __restrict__ idx, const float* __restrict__ wgt,
                    const int* __restrict__ countPtr) {
  const __bf16* xb = (const __bf16*)xb_;
  __bf16* act = (__bf16*)act_;
  const int cnt = countPtr ? *countPtr : T_;
  const int t0  = blockIdx.x * TM;
  if (t0 >= cnt) return;                    // expert tiles past routed count
  const int h0 = blockIdx.y * TN;

  __shared__ __bf16 As[2][TM * LSTR];       // double-buffered (async-filled)
  __shared__ __bf16 Bg[2][TN * LSTR];
  __shared__ __bf16 Bu[2][TN * LSTR];

  const int tid  = threadIdx.x;
  const int lane = tid & 31;
  const int wave = tid >> 5;
  const int waveM = (wave & 1) * 64;        // 2 wave rows of 64
  const int waveN = (wave >> 1) * 32;       // 4 wave cols of 32

  v8f accG[4][2], accU[4][2];
  const v8f z8 = {0.f, 0.f, 0.f, 0.f, 0.f, 0.f, 0.f, 0.f};
#pragma unroll
  for (int mi = 0; mi < 4; ++mi)
#pragma unroll
    for (int ni = 0; ni < 2; ++ni) { accG[mi][ni] = z8; accU[mi][ni] = z8; }

  // per-thread tile-loader coordinates: 128 rows x 32 cols, 16 elems/thread
  const int rowL = tid >> 1;
  const int colL = (tid & 1) * 16;

  const __bf16* arow = nullptr;             // gathered A row (nullptr => pad 0)
  {
    const int r = t0 + rowL;
    if (r < cnt) { const int t = idx ? idx[r] : r; arow = xb + (size_t)t * D_; }
  }
  const float* grow = wg + (size_t)(h0 + rowL) * D_ + colL;
  const float* urow = wu + (size_t)(h0 + rowL) * D_ + colL;

  __bf16* aDst0 = &As[0][rowL * LSTR + colL];
  __bf16* aDst1 = &As[1][rowL * LSTR + colL];
  if (!arow) {                              // pad rows: zero both buffers once
    const u32x4 z = {0, 0, 0, 0};
    ((u32x4*)aDst0)[0] = z; ((u32x4*)aDst0)[1] = z;
    ((u32x4*)aDst1)[0] = z; ((u32x4*)aDst1)[1] = z;
  }

  f32x4 gR[4], uR[4];

  // ---- stage tile 0 ----
  if (arow) {
    async_copy_b128(arow + colL,     aDst0);
    async_copy_b128(arow + colL + 8, aDst0 + 8);
  }
#pragma unroll
  for (int i = 0; i < 4; ++i) {
    gR[i] = ((const f32x4*)grow)[i];
    uR[i] = ((const f32x4*)urow)[i];
  }
  cvt_store16(&Bg[0][rowL * LSTR + colL], gR);
  cvt_store16(&Bu[0][rowL * LSTR + colL], uR);
  wait_async0();
  __syncthreads();

  for (int kb = 0; kb < D_; kb += TK) {
    const int  cur  = (kb / TK) & 1;
    const int  nxt  = cur ^ 1;
    const bool more = (kb + TK) < D_;

    if (more) {                             // stage tile k+1 while computing k
      if (arow) {
        __bf16* ad = nxt ? aDst1 : aDst0;
        async_copy_b128(arow + kb + TK + colL,     ad);
        async_copy_b128(arow + kb + TK + colL + 8, ad + 8);
      }
#pragma unroll
      for (int i = 0; i < 4; ++i) {
        gR[i] = ((const f32x4*)(grow + kb + TK))[i];
        uR[i] = ((const f32x4*)(urow + kb + TK))[i];
      }
    }

    // compute on tile k
    v16bf aF[4], gF[2], uF[2];
#pragma unroll
    for (int mi = 0; mi < 4; ++mi)
      aF[mi] = load_a_frag(&As[cur][0], waveM + mi * 16, lane);
#pragma unroll
    for (int ni = 0; ni < 2; ++ni) {
      gF[ni] = load_b_frag(&Bg[cur][0], waveN + ni * 16, lane);
      uF[ni] = load_b_frag(&Bu[cur][0], waveN + ni * 16, lane);
    }
#pragma unroll
    for (int mi = 0; mi < 4; ++mi)
#pragma unroll
      for (int ni = 0; ni < 2; ++ni) {
        accG[mi][ni] = wmma_bf16(aF[mi], gF[ni], accG[mi][ni]);
        accU[mi][ni] = wmma_bf16(aF[mi], uF[ni], accU[mi][ni]);
      }

    if (more) {
      cvt_store16(&Bg[nxt][rowL * LSTR + colL], gR);
      cvt_store16(&Bu[nxt][rowL * LSTR + colL], uR);
      wait_async0();                        // own async writes to buf[nxt] done
      __syncthreads();                      // publish buf[nxt]; allow reuse of cur
    }
  }

  // epilogue: silu(g) * u * combine-weight -> compacted bf16 activations
#pragma unroll
  for (int mi = 0; mi < 4; ++mi)
#pragma unroll
    for (int ni = 0; ni < 2; ++ni)
#pragma unroll
      for (int r8 = 0; r8 < 8; ++r8) {
        const int m = waveM + mi * 16 + r8 + (lane >> 4) * 8;  // C/D layout
        const int n = waveN + ni * 16 + (lane & 15);
        const int row = t0 + m;
        if (row < cnt) {
          const float g   = accG[mi][ni][r8];
          const float u   = accU[mi][ni][r8];
          const float sil = g / (1.0f + __expf(-g));
          const float w   = wgt ? wgt[row] : 1.0f;
          act[(size_t)row * H_ + h0 + n] = (__bf16)(sil * u * w);
        }
      }
}

// ---- gathered down-projection GEMM, TDM A staging, scatter into fp32 out ----
// out[idx[r], dcol] += act[r, :] . wd[dcol, :]   (K = H)
// A (compacted activations) is dense row-major -> one TENSOR_LOAD_TO_LDS per
// K-step stages the whole 128x32 tile, zero-filling rows past `cnt` via the
// descriptor's tensor_dim1 and writing the padded LSTR layout directly.
__global__ __launch_bounds__(256)
void down_kernel(const unsigned short* __restrict__ act_,
                 const float* __restrict__ wd, float* __restrict__ out,
                 const int* __restrict__ idx, const int* __restrict__ countPtr) {
  const __bf16* act = (const __bf16*)act_;
  const int cnt = countPtr ? *countPtr : T_;
  const int t0  = blockIdx.x * TM;
  if (t0 >= cnt) return;
  const int d0 = blockIdx.y * TN;

  __shared__ __bf16 As[2][TM * LSTR];
  __shared__ __bf16 Bs[2][TN * LSTR];

  const int tid  = threadIdx.x;
  const int lane = tid & 31;
  const int wave = tid >> 5;
  const int waveM = (wave & 1) * 64;
  const int waveN = (wave >> 1) * 32;

  v8f acc[4][2];
  const v8f z8 = {0.f, 0.f, 0.f, 0.f, 0.f, 0.f, 0.f, 0.f};
#pragma unroll
  for (int mi = 0; mi < 4; ++mi)
#pragma unroll
    for (int ni = 0; ni < 2; ++ni) acc[mi][ni] = z8;

  const int rowL = tid >> 1;
  const int colL = (tid & 1) * 16;
  const float* brow = wd + (size_t)(d0 + rowL) * H_ + colL;

  const unsigned rowsValid = (unsigned)(cnt - t0);       // OOB rows -> zeros
  const __bf16* atile = act + (size_t)t0 * H_;           // tile row 0
  const unsigned ldsA0 = (unsigned)(unsigned long long)&As[0][0];
  const unsigned ldsA1 = (unsigned)(unsigned long long)&As[1][0];

  f32x4 bR[4];

  // ---- stage tile 0 ----
  if (wave == 0)
    tdm_load_tile(ldsA0, atile, rowsValid, TM, H_);
#pragma unroll
  for (int i = 0; i < 4; ++i) bR[i] = ((const f32x4*)brow)[i];
  cvt_store16(&Bs[0][rowL * LSTR + colL], bR);
  if (wave == 0) __builtin_amdgcn_s_wait_tensorcnt(0);
  __syncthreads();

  for (int kb = 0; kb < H_; kb += TK) {
    const int  cur  = (kb / TK) & 1;
    const int  nxt  = cur ^ 1;
    const bool more = (kb + TK) < H_;

    if (more) {                             // stage tile k+1 while computing k
      if (wave == 0)
        tdm_load_tile(nxt ? ldsA1 : ldsA0, atile + kb + TK, rowsValid, TM, H_);
#pragma unroll
      for (int i = 0; i < 4; ++i) bR[i] = ((const f32x4*)(brow + kb + TK))[i];
    }

    v16bf aF[4], bF[2];
#pragma unroll
    for (int mi = 0; mi < 4; ++mi)
      aF[mi] = load_a_frag(&As[cur][0], waveM + mi * 16, lane);
#pragma unroll
    for (int ni = 0; ni < 2; ++ni)
      bF[ni] = load_b_frag(&Bs[cur][0], waveN + ni * 16, lane);
#pragma unroll
    for (int mi = 0; mi < 4; ++mi)
#pragma unroll
      for (int ni = 0; ni < 2; ++ni)
        acc[mi][ni] = wmma_bf16(aF[mi], bF[ni], acc[mi][ni]);

    if (more) {
      cvt_store16(&Bs[nxt][rowL * LSTR + colL], bR);
      if (wave == 0) __builtin_amdgcn_s_wait_tensorcnt(0);
      __syncthreads();
    }
  }

  // scatter-accumulate (rows unique within a kernel; kernels stream-serialized)
#pragma unroll
  for (int mi = 0; mi < 4; ++mi)
#pragma unroll
    for (int ni = 0; ni < 2; ++ni)
#pragma unroll
      for (int r8 = 0; r8 < 8; ++r8) {
        const int m = waveM + mi * 16 + r8 + (lane >> 4) * 8;
        const int n = waveN + ni * 16 + (lane & 15);
        const int row = t0 + m;
        if (row < cnt) {
          const int t = idx ? idx[row] : row;
          out[(size_t)t * D_ + d0 + n] += acc[mi][ni][r8];
        }
      }
}

// ---- host launcher ----------------------------------------------------------
extern "C" void kernel_launch(void* const* d_in, const int* in_sizes, int n_in,
                              void* d_out, int out_size, void* d_ws,
                              size_t ws_size, hipStream_t stream) {
  (void)in_sizes; (void)n_in; (void)out_size; (void)ws_size;
  const float* x   = (const float*)d_in[0];   // [B,S,D]
  const float* wr  = (const float*)d_in[1];   // [E,D]
  const float* shg = (const float*)d_in[2];   // [H,D]
  const float* shu = (const float*)d_in[3];   // [H,D]
  const float* shd = (const float*)d_in[4];   // [D,H]
  const float* exg = (const float*)d_in[5];   // [E,H,D]
  const float* exu = (const float*)d_in[6];   // [E,H,D]
  const float* exd = (const float*)d_in[7];   // [E,D,H]
  const float* ls  = (const float*)d_in[8];   // [D]
  // d_in[9] = top_k; fixed at 2 for this problem, hardcoded in router_kernel.

  // workspace layout (~49 MB)
  char* ws = (char*)d_ws;
  size_t off = 0;
  int*  counts = (int*)(ws + off);            off += 256;                        // E ints (padded)
  int*  idx    = (int*)(ws + off);            off += (size_t)E_ * T_ * 4;        // gather lists
  float* wgt   = (float*)(ws + off);          off += (size_t)E_ * T_ * 4;        // combine weights
  unsigned short* xb  = (unsigned short*)(ws + off); off += (size_t)T_ * D_ * 2; // x in bf16
  unsigned short* act = (unsigned short*)(ws + off);                             // [T,H] bf16
  float* out = (float*)d_out;

  zero_f32_kernel<<<2048, 256, 0, stream>>>(out, (size_t)T_ * D_);
  zero_i32_kernel<<<1, 32, 0, stream>>>(counts, E_);
  router_kernel<<<T_ / 8, 256, 0, stream>>>(x, wr, counts, idx, wgt);
  cvt_bf16_kernel<<<2048, 256, 0, stream>>>(x, xb, (size_t)T_ * D_);

  const dim3 gAct(T_ / TM, H_ / TN);   // 64 x 16
  const dim3 gDown(T_ / TM, D_ / TN);  // 64 x 8

  // shared expert: identity gather, weight 1
  gate_up_kernel<<<gAct, 256, 0, stream>>>(xb, shg, shu, act, nullptr, nullptr, nullptr);
  down_kernel<<<gDown, 256, 0, stream>>>(act, shd, out, nullptr, nullptr);

  // routed experts: gathered tiles, blocks past the routed count exit early
  for (int e = 0; e < E_; ++e) {
    gate_up_kernel<<<gAct, 256, 0, stream>>>(
        xb, exg + (size_t)e * H_ * D_, exu + (size_t)e * H_ * D_, act,
        idx + (size_t)e * T_, wgt + (size_t)e * T_, counts + e);
    down_kernel<<<gDown, 256, 0, stream>>>(
        act, exd + (size_t)e * D_ * H_, out, idx + (size_t)e * T_, counts + e);
  }

  scale_kernel<<<2048, 256, 0, stream>>>(out, ls, (size_t)T_ * D_);
}
